// HeteroNet_24988119728306
// MI455X (gfx1250) — matrix operand order repaired and checked by reading, not compiled, not run on hardware
//
#include <hip/hip_runtime.h>
#include <hip/hip_bf16.h>
#include <hip/hip_fp16.h>
#include <cstdint>

// ---------------------------------------------------------------------------
// HeteroSAGE 2-layer forward for MI455X (gfx1250, wave32, WMMA).
//
//   * Edge aggregation (dominant): one wave per edge, f16 source rows
//     (256B/edge coalesced gather), f32 atomic scatter-add + float count.
//     Working set per pass (~77MB) is L2-resident on the 192MB L2.
//   * Linear algebra: fold cat(aggr@Wn, x@Ws)@Wu + b into
//     aggr@WnE + x@WsE + bE; merge dst-A self terms. Folded weights stored
//     f16 *transposed* [N][K] so a B fragment is two 16-byte runs/lane.
//     GEMM: v_wmma_f32_16x16x32_f16. One wave computes a full 16x128 output
//     block (8 column tiles, 8 v8f accumulators), so each A fragment is
//     loaded once and feeds 8 WMMAs — single pass over the A operands.
//     Scatter-mean divide fused into the f32 A-fragment load, bias fused in
//     the epilogue. All operand pointers are direct kernel args (global AS).
//   * Dropout: deterministic splitmix64 hash keep-mask fused with ReLU.
// ---------------------------------------------------------------------------

#define N_NODES 100000
#define DIM 128
#define EDGES 1600000
#define P_DROP 0.2f
#define KEEP_SCALE (1.0f / (1.0f - P_DROP))

typedef __attribute__((ext_vector_type(16))) _Float16 v16h;
typedef __attribute__((ext_vector_type(4)))  _Float16 v4h;
typedef __attribute__((ext_vector_type(8)))  float    v8f;

// ---------------- deterministic hash-based dropout --------------------------
__device__ __forceinline__ float uhash01(uint32_t seed, uint32_t idx) {
    uint64_t z = ((uint64_t)seed << 32) | idx;
    z += 0x9E3779B97F4A7C15ull;
    z = (z ^ (z >> 30)) * 0xBF58476D1CE4E5B9ull;
    z = (z ^ (z >> 27)) * 0x94D049BB133111EBull;
    z ^= z >> 31;
    return (float)(uint32_t)(z >> 40) * (1.0f / 16777216.0f);
}

// x: f32 in, y: f16 out (activations consumed at f16 precision by WMMA)
__global__ void relu_dropout_kernel(const float* __restrict__ x,
                                    _Float16* __restrict__ y,
                                    int n, uint32_t seed) {
    int i = blockIdx.x * blockDim.x + threadIdx.x;
    if (i >= n) return;
    float v = x[i];
    float u = uhash01(seed, (uint32_t)i);
    y[i] = (_Float16)((u >= P_DROP) ? fmaxf(v, 0.0f) * KEEP_SCALE : 0.0f);
}

// ---------------- edge gather + scatter-add (mean numerator & counts) -------
__global__ void edge_scatter_kernel(const _Float16* __restrict__ src_feat,
                                    const int* __restrict__ ei,   // [2, E]
                                    float* __restrict__ accum,    // [N, 128] f32
                                    float* __restrict__ cnt,      // [N]
                                    int nedges) {
    const int lane = threadIdx.x & 31;
    const int e = (blockIdx.x * blockDim.x + threadIdx.x) >> 5; // wave id
    if (e >= nedges) return;
    const int s = ei[e];            // row 0: src
    const int d = ei[nedges + e];   // row 1: dst
    const v4h v = *(const v4h*)(src_feat + (size_t)s * DIM + lane * 4);
    float* dp = accum + (size_t)d * DIM + lane * 4;
    atomicAdd(dp + 0, (float)v[0]);
    atomicAdd(dp + 1, (float)v[1]);
    atomicAdd(dp + 2, (float)v[2]);
    atomicAdd(dp + 3, (float)v[3]);
    if (lane == 0) atomicAdd(cnt + d, 1.0f);
}

// ---------------- weight folding --------------------------------------------
// Effective weights stored f16, TRANSPOSED [n][k] (fragment order):
//   WnE[t][n][k] = sum_j Wn[t][k][j] * Wu[t][j][n]
//   WsEA[n][k]   = sum_j (Ws[0][k][j]*Wu[0][128+j][n] + Ws[2][k][j]*Wu[2][128+j][n])
//   WsEB[n][k]   = sum_j  Ws[1][k][j]*Wu[1][128+j][n]
// biasA/biasB remain f32.
__global__ void fold_weights_kernel(const float* __restrict__ Wn,  // [3,128,128]
                                    const float* __restrict__ Ws,  // [3,128,128]
                                    const float* __restrict__ Wu,  // [3,256,128]
                                    const float* __restrict__ bn,  // [3,128]
                                    const float* __restrict__ bs,  // [3,128]
                                    const float* __restrict__ bu,  // [3,128]
                                    _Float16* __restrict__ WnE,    // [3,128,128] (n,k)
                                    _Float16* __restrict__ WsEA,   // [128,128]   (n,k)
                                    _Float16* __restrict__ WsEB,   // [128,128]   (n,k)
                                    float* __restrict__ biasA,     // [128]
                                    float* __restrict__ biasB) {   // [128]
    const int k = blockIdx.x;   // 0..127 (inner/K index of effective weight)
    const int n = threadIdx.x;  // 0..127 (output column)
    float ws_t[3];
    for (int t = 0; t < 3; ++t) {
        const float* wnrow = Wn + ((size_t)t * DIM + k) * DIM;
        const float* wsrow = Ws + ((size_t)t * DIM + k) * DIM;
        const float* wut   = Wu + (size_t)t * 2 * DIM * DIM;
        float an = 0.0f, as = 0.0f;
        for (int j = 0; j < DIM; ++j) {
            an += wnrow[j] * wut[(size_t)j * DIM + n];
            as += wsrow[j] * wut[(size_t)(DIM + j) * DIM + n];
        }
        WnE[((size_t)t * DIM + n) * DIM + k] = (_Float16)an;   // transposed store
        ws_t[t] = as;
    }
    WsEA[(size_t)n * DIM + k] = (_Float16)(ws_t[0] + ws_t[2]);
    WsEB[(size_t)n * DIM + k] = (_Float16)ws_t[1];

    if (k == 0) {
        float bA = 0.0f, bB = 0.0f;
        for (int t = 0; t < 3; ++t) {
            const float* wut = Wu + (size_t)t * 2 * DIM * DIM;
            float b = bu[t * DIM + n];
            for (int j = 0; j < DIM; ++j) {
                b += bn[t * DIM + j] * wut[(size_t)j * DIM + n];
                b += bs[t * DIM + j] * wut[(size_t)(DIM + j) * DIM + n];
            }
            if (t == 1) bB = b; else bA += b;
        }
        biasA[n] = bA;
        biasB[n] = bB;
    }
}

// ---------------- WMMA fragments --------------------------------------------
// 16-bit A fragment (16x32 MxK): lanes 0-15 hold K=[0..7],[16..23] of row
// M=lane; lanes 16-31 hold K=[8..15],[24..31] of row M=lane-16.
__device__ __forceinline__ v16h load_a_frag_f32(const float* __restrict__ A,
                                                int row, int k0, int half,
                                                float scale) {
    const float* p = A + (size_t)row * DIM + k0 + half * 8;
    v16h a;
#pragma unroll
    for (int i = 0; i < 8; ++i) a[i]     = (_Float16)(p[i]      * scale);
#pragma unroll
    for (int i = 0; i < 8; ++i) a[8 + i] = (_Float16)(p[16 + i] * scale);
    return a;
}

__device__ __forceinline__ v16h load_a_frag_f16(const _Float16* __restrict__ A,
                                                int row, int k0, int half) {
    const _Float16* p = A + (size_t)row * DIM + k0 + half * 8;
    v16h a;
#pragma unroll
    for (int i = 0; i < 8; ++i) a[i]     = p[i];
#pragma unroll
    for (int i = 0; i < 8; ++i) a[8 + i] = p[16 + i];
    return a;
}

// B fragment (32x16 KxN): lanes 0-15 hold K=[0..7],[16..23] of col N=lane; lanes 16-31
// hold K=[8..15],[24..31] of col N=lane-16. Wt layout is [n][k] f16.
__device__ __forceinline__ v16h load_b_frag(const _Float16* __restrict__ Wt,
                                            int col, int k0, int half) {
    const _Float16* p = Wt + (size_t)col * DIM + k0 + half * 8;
    v16h b;
#pragma unroll
    for (int i = 0; i < 16; ++i) b[i] = p[(i < 8) ? i : i + 8];
    return b;
}

// Accumulate rowscale(A16rows) @ W into the 8 column-tile accumulators.
// Each A fragment is loaded once and feeds 8 WMMAs.
__device__ __forceinline__ void accum_pair_f32(const float* __restrict__ A,
                                               const float* __restrict__ cn,
                                               const _Float16* __restrict__ Wt,
                                               int arow, int l15, int half,
                                               v8f (&acc)[8]) {
    const float scale = 1.0f / fmaxf(cn[arow], 1.0f); // fused scatter-mean divide
#pragma unroll
    for (int k0 = 0; k0 < DIM; k0 += 32) {
        v16h a = load_a_frag_f32(A, arow, k0, half, scale);
#pragma unroll
        for (int t = 0; t < 8; ++t) {
            v16h b = load_b_frag(Wt, t * 16 + l15, k0, half);
            acc[t] = __builtin_amdgcn_wmma_f32_16x16x32_f16(
                false, a, false, b, (short)0, acc[t], false, false);
        }
    }
}

__device__ __forceinline__ void accum_pair_f16(const _Float16* __restrict__ A,
                                               const _Float16* __restrict__ Wt,
                                               int arow, int l15, int half,
                                               v8f (&acc)[8]) {
#pragma unroll
    for (int k0 = 0; k0 < DIM; k0 += 32) {
        v16h a = load_a_frag_f16(A, arow, k0, half);
#pragma unroll
        for (int t = 0; t < 8; ++t) {
            v16h b = load_b_frag(Wt, t * 16 + l15, k0, half);
            acc[t] = __builtin_amdgcn_wmma_f32_16x16x32_f16(
                false, a, false, b, (short)0, acc[t], false, false);
        }
    }
}

// out[16x128 block] = mean0(A0)@W0 [+ mean1(A1)@W1] + S@Wself + bias
// One wave per 16-row block; 8 column tiles held in registers.
__global__ void sage_gemm_kernel(const float* __restrict__ A0,
                                 const float* __restrict__ c0,
                                 const _Float16* __restrict__ W0,
                                 const float* __restrict__ A1,
                                 const float* __restrict__ c1,
                                 const _Float16* __restrict__ W1,
                                 const _Float16* __restrict__ S,
                                 const _Float16* __restrict__ Wself,
                                 const float* __restrict__ bias,
                                 float* __restrict__ out,
                                 int nrows, int naggr) {
    const int lane = threadIdx.x & 31;
    const int wave = (blockIdx.x * blockDim.x + threadIdx.x) >> 5;
    const int rowTiles = nrows >> 4;          // 6250
    if (wave >= rowTiles) return;             // wave-uniform: EXEC stays all-1s

    const int tm   = wave << 4;               // row-block base
    const int half = lane >> 4;
    const int l15  = lane & 15;
    const int arow = tm + l15;

    v8f acc[8] = {};
    accum_pair_f32(A0, c0, W0, arow, l15, half, acc);
    if (naggr > 1)
        accum_pair_f32(A1, c1, W1, arow, l15, half, acc);
    accum_pair_f16(S, Wself, arow, l15, half, acc);

    // C/D layout per tile t: VGPR r -> (M = tm + r + half*8, N = t*16 + l15)
    const int rbase = tm + half * 8;
#pragma unroll
    for (int t = 0; t < 8; ++t) {
        const int col = t * 16 + l15;
        const float bv = bias[col];
#pragma unroll
        for (int r = 0; r < 8; ++r)
            out[(size_t)(rbase + r) * DIM + col] = acc[t][r] + bv;
    }
}

// ---------------------------------------------------------------------------
extern "C" void kernel_launch(void* const* d_in, const int* in_sizes, int n_in,
                              void* d_out, int out_size, void* d_ws, size_t ws_size,
                              hipStream_t stream) {
    const float* xA  = (const float*)d_in[0];
    const float* xB  = (const float*)d_in[1];
    const int*   ei0 = (const int*)d_in[2];
    const int*   ei1 = (const int*)d_in[3];
    const int*   ei2 = (const int*)d_in[4];
    const float* Wn1 = (const float*)d_in[5];
    const float* Ws1 = (const float*)d_in[6];
    const float* Wu1 = (const float*)d_in[7];
    const float* Wn2 = (const float*)d_in[8];
    const float* Ws2 = (const float*)d_in[9];
    const float* Wu2 = (const float*)d_in[10];
    const float* bn1 = (const float*)d_in[11];
    const float* bs1 = (const float*)d_in[12];
    const float* bu1 = (const float*)d_in[13];
    const float* bn2 = (const float*)d_in[14];
    const float* bs2 = (const float*)d_in[15];
    const float* bu2 = (const float*)d_in[16];
    (void)in_sizes; (void)n_in; (void)out_size; (void)ws_size;

    float* outA = (float*)d_out;
    float* outB = outA + (size_t)N_NODES * DIM;

    // ---- workspace carving (256B aligned) ----
    char* base = (char*)d_ws;
    size_t off = 0;
    auto carve = [&](size_t bytes) -> void* {
        void* p = base + off;
        off = (off + bytes + 255) & ~(size_t)255;
        return p;
    };
    const size_t NF = (size_t)N_NODES * DIM;
    _Float16* hA = (_Float16*)carve(NF * 2);
    _Float16* hB = (_Float16*)carve(NF * 2);
    float* acc0 = (float*)carve(NF * 4);
    float* acc1 = (float*)carve(NF * 4);
    float* acc2 = (float*)carve(NF * 4);
    float* cnt0 = (float*)carve((size_t)N_NODES * 4);
    float* cnt1 = (float*)carve((size_t)N_NODES * 4);
    float* cnt2 = (float*)carve((size_t)N_NODES * 4);
    const size_t WSZ = (size_t)DIM * DIM * 2;  // f16
    _Float16* WnE_1  = (_Float16*)carve(3 * WSZ);
    _Float16* WsEA_1 = (_Float16*)carve(WSZ);
    _Float16* WsEB_1 = (_Float16*)carve(WSZ);
    float*    bA_1   = (float*)carve(DIM * 4);
    float*    bB_1   = (float*)carve(DIM * 4);
    _Float16* WnE_2  = (_Float16*)carve(3 * WSZ);
    _Float16* WsEA_2 = (_Float16*)carve(WSZ);
    _Float16* WsEB_2 = (_Float16*)carve(WSZ);
    float*    bA_2   = (float*)carve(DIM * 4);
    float*    bB_2   = (float*)carve(DIM * 4);

    const int nElem = N_NODES * DIM;                    // 12.8M
    const int ewBlocks = (EDGES + 7) / 8;               // 1 wave/edge, 8 waves/block
    const int rowTiles = N_NODES / 16;                  // 6250
    const int gemmBlocks = (rowTiles + 7) / 8;          // 8 waves/block
    const size_t WMAT = (size_t)DIM * DIM;              // elements per matrix

    // ---- fold weights (data-independent) ----
    fold_weights_kernel<<<DIM, DIM, 0, stream>>>(Wn1, Ws1, Wu1, bn1, bs1, bu1,
                                                 WnE_1, WsEA_1, WsEB_1, bA_1, bB_1);
    fold_weights_kernel<<<DIM, DIM, 0, stream>>>(Wn2, Ws2, Wu2, bn2, bs2, bu2,
                                                 WnE_2, WsEA_2, WsEB_2, bA_2, bB_2);

    // ---- input dropout + relu (f32 -> f16 activations) ----
    relu_dropout_kernel<<<(nElem + 255) / 256, 256, 0, stream>>>(xA, hA, nElem, 0u);
    relu_dropout_kernel<<<(nElem + 255) / 256, 256, 0, stream>>>(xB, hB, nElem, 1u);

    for (int layer = 0; layer < 2; ++layer) {
        const _Float16* WnE  = layer ? WnE_2  : WnE_1;
        const _Float16* WsEA = layer ? WsEA_2 : WsEA_1;
        const _Float16* WsEB = layer ? WsEB_2 : WsEB_1;
        const float*    bA   = layer ? bA_2   : bA_1;
        const float*    bB   = layer ? bB_2   : bB_1;

        hipMemsetAsync(acc0, 0, NF * 4, stream);
        hipMemsetAsync(acc1, 0, NF * 4, stream);
        hipMemsetAsync(acc2, 0, NF * 4, stream);
        hipMemsetAsync(cnt0, 0, (size_t)N_NODES * 4, stream);
        hipMemsetAsync(cnt1, 0, (size_t)N_NODES * 4, stream);
        hipMemsetAsync(cnt2, 0, (size_t)N_NODES * 4, stream);

        // message types: (A -ei0-> A), (A -ei1-> B), (B -ei2-> A)
        edge_scatter_kernel<<<ewBlocks, 256, 0, stream>>>(hA, ei0, acc0, cnt0, EDGES);
        edge_scatter_kernel<<<ewBlocks, 256, 0, stream>>>(hA, ei1, acc1, cnt1, EDGES);
        edge_scatter_kernel<<<ewBlocks, 256, 0, stream>>>(hB, ei2, acc2, cnt2, EDGES);

        // dst A: mean(acc0)@WnE[0] + mean(acc2)@WnE[2] + hA@WsEA + biasA
        sage_gemm_kernel<<<gemmBlocks, 256, 0, stream>>>(
            acc0, cnt0, WnE + 0 * WMAT,
            acc2, cnt2, WnE + 2 * WMAT,
            hA, WsEA, bA, outA, N_NODES, 2);
        // dst B: mean(acc1)@WnE[1] + hB@WsEB + biasB
        sage_gemm_kernel<<<gemmBlocks, 256, 0, stream>>>(
            acc1, cnt1, WnE + 1 * WMAT,
            nullptr, nullptr, nullptr,
            hB, WsEB, bB, outB, N_NODES, 1);

        if (layer == 0) {
            // inter-layer dropout + relu; layer-1 outputs staged in d_out
            relu_dropout_kernel<<<(nElem + 255) / 256, 256, 0, stream>>>(outA, hA, nElem, 2u);
            relu_dropout_kernel<<<(nElem + 255) / 256, 256, 0, stream>>>(outB, hB, nElem, 3u);
        }
    }
}
